// LiquidNCPNetwork_4552665334408
// MI455X (gfx1250) — matrix operand order repaired and checked by reading, not compiled, not run on hardware
//
#include <hip/hip_runtime.h>
#include <math.h>

// ---------------- problem constants (mirror reference) ----------------
#define BB      64
#define SS      32
#define INF     512
#define INTER   1229
#define CMD     819
#define MOTOR   512
#define NUNITS  2560

// head dims (K of each GEMM) and padded sizes
#define H1 1741
#define H2 2048
#define H3 1331
#define K1 1760   // pad to mult of 32
#define K2 2048
#define K3 1344
#define N1 1232   // pad n_h to mult of 16
#define N2 832
#define N3 512

// workspace layout in __bf16 elements
#define WT1_OFF 0
#define WT1_SZ  (4*N1*K1)
#define WT2_OFF (WT1_OFF + WT1_SZ)
#define WT2_SZ  (4*N2*K2)
#define WT3_OFF (WT2_OFF + WT2_SZ)
#define WT3_SZ  (4*N3*K3)
#define Z1_OFF  (WT3_OFF + WT3_SZ)
#define Z1_SZ   (2*BB*K1)
#define Z2_OFF  (Z1_OFF + Z1_SZ)
#define Z2_SZ   (2*BB*K2)
#define Z3_OFF  (Z2_OFF + Z2_SZ)
#define Z3_SZ   (2*BB*K3)
#define ZTOT    (Z1_SZ + Z2_SZ + Z3_SZ)

typedef __bf16 v8bf  __attribute__((ext_vector_type(8)));
typedef __bf16 v16bf __attribute__((ext_vector_type(16)));
typedef float  v8f   __attribute__((ext_vector_type(8)));

#define LDS_ROW   40                    // 32 data + 8 pad bf16 -> 80B stride, bank-friendly
#define BUF_ELEMS (4 * 16 * LDS_ROW)    // one double-buffer slot (bf16 elems)

// ---------------- weight prep: dst[head][n][k] = (bf16)(W[n][k]*mask[n][k]) ----------------
__global__ void prep_w(const float* __restrict__ W0, const float* __restrict__ W1,
                       const float* __restrict__ W2, const float* __restrict__ W3,
                       const float* __restrict__ mask, __bf16* __restrict__ dst,
                       int n_h, int head, int Kpad, int Npad) {
    long idx = (long)blockIdx.x * blockDim.x + threadIdx.x;
    long per = (long)Npad * Kpad;
    long tot = 4 * per;
    if (idx >= tot) return;
    int hd = (int)(idx / per);
    long rem = idx - (long)hd * per;
    int n = (int)(rem / Kpad);
    int k = (int)(rem - (long)n * Kpad);
    const float* W = (hd == 0) ? W0 : (hd == 1) ? W1 : (hd == 2) ? W2 : W3;
    float v = 0.0f;
    if (n < n_h && k < head) {
        long e = (long)n * head + k;
        v = W[e] * mask[e];
    }
    dst[idx] = (__bf16)v;
}

__global__ void zero_bf(__bf16* __restrict__ dst, int count) {
    int i = blockIdx.x * blockDim.x + threadIdx.x;
    if (i < count) dst[i] = (__bf16)0.0f;
}

// scatter h0 (B, NUNITS) f32 into the parity-0 z buffers (bf16)
__global__ void init_h(const float* __restrict__ h0,
                       __bf16* __restrict__ z1, __bf16* __restrict__ z2, __bf16* __restrict__ z3) {
    int i = blockIdx.x * blockDim.x + threadIdx.x;
    if (i >= BB * NUNITS) return;
    int m = i / NUNITS, u = i - m * NUNITS;
    float v = h0[i];
    if (u < INTER)            z1[(long)m * K1 + INF + u]              = (__bf16)v;
    else if (u < INTER + CMD) z2[(long)m * K2 + INTER + (u - INTER)]  = (__bf16)v;
    else                      z3[(long)m * K3 + CMD + (u - INTER - CMD)] = (__bf16)v;
}

// pack x[:, t, :] (f32) into z1[parity] columns [0,512)
__global__ void pack_x(const float* __restrict__ x, __bf16* __restrict__ z1, int t) {
    int i = blockIdx.x * blockDim.x + threadIdx.x;
    if (i >= BB * INF) return;
    int m = i / INF, k = i - m * INF;
    z1[(long)m * K1 + k] = (__bf16)x[((long)m * SS + t) * INF + k];
}

// ---------------- fused 4-head WMMA GEMM + CfC epilogue ----------------
// grid.x = Npad/16 column tiles; block = 128 (4 waves, one batch row-tile of 16 each).
// B tiles stream L2 -> LDS via GLOBAL_LOAD_ASYNC_TO_LDS_B128 (ASYNCcnt), double-buffered.
__global__ __launch_bounds__(128)
void lncp_cell(const __bf16* __restrict__ A, int Kpad,
               const __bf16* __restrict__ WT, int headStride,
               const float* __restrict__ bg,  const float* __restrict__ bh,
               const float* __restrict__ bfg, const float* __restrict__ bfh,
               int Nvalid,
               const float* __restrict__ tsPtr, int t,
               __bf16* __restrict__ dstA, int strideA, int colA,
               __bf16* __restrict__ dstB, int strideB, int colB,
               float* __restrict__ hf, int hfcol,
               float* __restrict__ y) {
    __shared__ __bf16 ldsB[2 * BUF_ELEMS];   // double-buffered; only shared var -> LDS offset 0

    const int n0   = blockIdx.x << 4;
    const int wave = threadIdx.x >> 5;
    const int lane = threadIdx.x & 31;
    const int lhalf = lane >> 4;           // 0: lanes 0-15, 1: lanes 16-31
    const int l15   = lane & 15;

    // per-thread staging assignment: chunks q and q+128 of 256 16-byte chunks
    const int q0 = threadIdx.x;
    const int hd0 = q0 >> 6,        rem0 = q0 & 63;
    const int nn0 = rem0 >> 2,      ch0  = rem0 & 3;
    const int q1 = q0 + 128;
    const int hd1 = q1 >> 6,        rem1 = q1 & 63;
    const int nn1 = rem1 >> 2,      ch1  = rem1 & 3;
    const __bf16* g0 = WT + (long)hd0 * headStride + (long)(n0 + nn0) * Kpad + ch0 * 8;
    const __bf16* g1 = WT + (long)hd1 * headStride + (long)(n0 + nn1) * Kpad + ch1 * 8;
    const int l0e = (hd0 * 16 + nn0) * LDS_ROW + ch0 * 8;   // element index in one buffer
    const int l1e = (hd1 * 16 + nn1) * LDS_ROW + ch1 * 8;

    v8f acc0 = {}, acc1 = {}, acc2 = {}, acc3 = {};

    const int mrow = (wave << 4) + l15;    // A-matrix row for this lane
    const __bf16* Arow = A + (long)mrow * Kpad;

    // Async fill of the two 16B chunks this thread owns, K-slice kk, buffer slot.
    // The LDS destination address is derived from &ldsB[...] (generic ptr truncated to
    // the 32-bit LDS offset) so ldsB escapes into the asm: the compiler must assume the
    // asm writes it, keeping the subsequent ds_load_b128 reads live and ordered.
    #define STAGE(kk_, buf_)                                                          \
        do {                                                                          \
            unsigned long long a0_ = (unsigned long long)(g0 + (kk_));                \
            unsigned long long a1_ = (unsigned long long)(g1 + (kk_));                \
            unsigned d0_ = (unsigned)(unsigned long long)                             \
                           (ldsB + (buf_) * BUF_ELEMS + l0e);                         \
            unsigned d1_ = (unsigned)(unsigned long long)                             \
                           (ldsB + (buf_) * BUF_ELEMS + l1e);                         \
            asm volatile("global_load_async_to_lds_b128 %0, %1, off"                  \
                         :: "v"(d0_), "v"(a0_) : "memory");                           \
            asm volatile("global_load_async_to_lds_b128 %0, %1, off"                  \
                         :: "v"(d1_), "v"(a1_) : "memory");                           \
        } while (0)

    // prologue: fill buffer 0
    STAGE(0, 0);
    asm volatile("s_wait_asynccnt 0x0" ::: "memory");
    __syncthreads();

    int buf = 0;
    for (int kk = 0; kk < Kpad; kk += 32) {
        // prefetch next K-slice into the other buffer (reads of it finished last iter)
        if (kk + 32 < Kpad) STAGE(kk + 32, buf ^ 1);

        // ---- A fragment: lane m=l15; K chunks at {0,+16}, high lanes shifted +8 ----
        int kofs = kk + (lhalf << 3);
        v8bf alo = *(const v8bf*)(Arow + kofs);
        v8bf ahi = *(const v8bf*)(Arow + kofs + 16);
        v16bf afrag = __builtin_shufflevector(alo, ahi,
            0,1,2,3,4,5,6,7,8,9,10,11,12,13,14,15);

        // ---- B fragments from LDS: col n=l15; K chunks at {0,+16}, high lanes shifted +8 ----
        const __bf16* base = ldsB + buf * BUF_ELEMS;
        #pragma unroll
        for (int hd = 0; hd < 4; ++hd) {
            const __bf16* bp = base + (hd * 16 + l15) * LDS_ROW + (lhalf << 3);
            v8bf blo = *(const v8bf*)bp;
            v8bf bhi = *(const v8bf*)(bp + 16);
            v16bf bfrag = __builtin_shufflevector(blo, bhi,
                0,1,2,3,4,5,6,7,8,9,10,11,12,13,14,15);
            if (hd == 0)      acc0 = __builtin_amdgcn_wmma_f32_16x16x32_bf16(false, afrag, false, bfrag, (short)0, acc0, false, false);
            else if (hd == 1) acc1 = __builtin_amdgcn_wmma_f32_16x16x32_bf16(false, afrag, false, bfrag, (short)0, acc1, false, false);
            else if (hd == 2) acc2 = __builtin_amdgcn_wmma_f32_16x16x32_bf16(false, afrag, false, bfrag, (short)0, acc2, false, false);
            else              acc3 = __builtin_amdgcn_wmma_f32_16x16x32_bf16(false, afrag, false, bfrag, (short)0, acc3, false, false);
        }

        // next buffer complete + all waves done reading current -> single barrier
        asm volatile("s_wait_asynccnt 0x0" ::: "memory");
        __syncthreads();
        buf ^= 1;
    }
    #undef STAGE

    // ---- fused CfC epilogue: o = tanh(g)*(1-gate) + tanh(h)*gate ----
    const int n = n0 + l15;
    const bool valid = (n < Nvalid);
    const float ts = tsPtr[t];
    const float vbg  = valid ? bg[n]  : 0.0f;
    const float vbh  = valid ? bh[n]  : 0.0f;
    const float vbfg = valid ? bfg[n] : 0.0f;
    const float vbfh = valid ? bfh[n] : 0.0f;

    #pragma unroll
    for (int j = 0; j < 8; ++j) {
        int m = (wave << 4) + j + (lhalf << 3);       // C/D: row = j (+8 high lanes)
        float g    = tanhf(acc0[j] + vbg);
        float hh   = tanhf(acc1[j] + vbh);
        float pre  = (acc2[j] + vbfg) + ts * (acc3[j] + vbfh);
        float gate = 1.0f / (1.0f + expf(-pre));
        float o    = g * (1.0f - gate) + hh * gate;
        if (valid) {
            dstA[(long)m * strideA + colA + n] = (__bf16)o;
            if (dstB) dstB[(long)m * strideB + colB + n] = (__bf16)o;
            hf[(long)m * NUNITS + hfcol + n] = o;
            if (y) y[((long)m * SS + t) * MOTOR + n] = o;
        }
    }
}

// ---------------- host-side orchestration ----------------
extern "C" void kernel_launch(void* const* d_in, const int* in_sizes, int n_in,
                              void* d_out, int out_size, void* d_ws, size_t ws_size,
                              hipStream_t stream) {
    const float* x  = (const float*)d_in[0];
    const float* h0 = (const float*)d_in[1];
    const float* ts = (const float*)d_in[2];

    // per-cell params: base 3 + 9*(cell-1): Wg, Wh, Wfg, Wfh, bg, bh, bfg, bfh, mask
    const float* P[3][9];
    for (int c = 0; c < 3; ++c)
        for (int j = 0; j < 9; ++j)
            P[c][j] = (const float*)d_in[3 + 9 * c + j];

    __bf16* ws  = (__bf16*)d_ws;
    __bf16* wt1 = ws + WT1_OFF;
    __bf16* wt2 = ws + WT2_OFF;
    __bf16* wt3 = ws + WT3_OFF;
    __bf16* z1b = ws + Z1_OFF;   // two parities, each BB*K1
    __bf16* z2b = ws + Z2_OFF;
    __bf16* z3b = ws + Z3_OFF;

    float* yout = (float*)d_out;                          // (B, S, MOTOR)
    float* hf   = (float*)d_out + (long)BB * SS * MOTOR;  // (B, NUNITS)

    // 1) fold mask into weights, bf16, padded (row-major n x Kpad) -> L2-resident
    {
        long t1 = (long)WT1_SZ, t2 = (long)WT2_SZ, t3 = (long)WT3_SZ;
        prep_w<<<dim3((unsigned)((t1 + 255) / 256)), 256, 0, stream>>>(
            P[0][0], P[0][1], P[0][2], P[0][3], P[0][8], wt1, INTER, H1, K1, N1);
        prep_w<<<dim3((unsigned)((t2 + 255) / 256)), 256, 0, stream>>>(
            P[1][0], P[1][1], P[1][2], P[1][3], P[1][8], wt2, CMD, H2, K2, N2);
        prep_w<<<dim3((unsigned)((t3 + 255) / 256)), 256, 0, stream>>>(
            P[2][0], P[2][1], P[2][2], P[2][3], P[2][8], wt3, MOTOR, H3, K3, N3);
    }
    // 2) zero z-staging (covers K-padding columns), then seed h0 into parity-0 buffers
    zero_bf<<<dim3((ZTOT + 255) / 256), 256, 0, stream>>>(z1b, ZTOT);
    init_h<<<dim3((BB * NUNITS + 255) / 256), 256, 0, stream>>>(h0, z1b, z2b, z3b);

    // 3) sequential scan: 3 dependent fused GEMM+activation kernels per step
    for (int t = 0; t < SS; ++t) {
        int p = t & 1, q = 1 - p;
        __bf16* z1p = z1b + (long)p * BB * K1;  __bf16* z1q = z1b + (long)q * BB * K1;
        __bf16* z2p = z2b + (long)p * BB * K2;  __bf16* z2q = z2b + (long)q * BB * K2;
        __bf16* z3p = z3b + (long)p * BB * K3;  __bf16* z3q = z3b + (long)q * BB * K3;

        pack_x<<<dim3((BB * INF + 255) / 256), 256, 0, stream>>>(x, z1p, t);

        // cell1: A=[x_t | h1] -> o1; o1 feeds z2[p][:, :INTER] and z1[q][:, INF:]
        lncp_cell<<<dim3(N1 / 16), 128, 0, stream>>>(
            z1p, K1, wt1, N1 * K1,
            P[0][4], P[0][5], P[0][6], P[0][7], INTER, ts, t,
            z2p, K2, 0,
            z1q, K1, INF,
            hf, 0, nullptr);

        // cell2: A=[o1 | h2] -> o2; o2 feeds z3[p][:, :CMD] and z2[q][:, INTER:]
        lncp_cell<<<dim3(N2 / 16), 128, 0, stream>>>(
            z2p, K2, wt2, N2 * K2,
            P[1][4], P[1][5], P[1][6], P[1][7], CMD, ts, t,
            z3p, K3, 0,
            z2q, K2, INTER,
            hf, INTER, nullptr);

        // cell3: A=[o2 | h3] -> o3; o3 -> y_pred[:, t, :], z3[q][:, CMD:], h_final tail
        lncp_cell<<<dim3(N3 / 16), 128, 0, stream>>>(
            z3p, K3, wt3, N3 * K3,
            P[2][4], P[2][5], P[2][6], P[2][7], MOTOR, ts, t,
            z3q, K3, CMD,
            nullptr, 0, 0,
            hf, INTER + CMD, yout);
    }
}
